// MultiHeadSelfAttention_15857019257243
// MI455X (gfx1250) — compile-verified, matrix operand order checked by reading, not run on hardware
//
#include <hip/hip_runtime.h>
#include <hip/hip_bf16.h>
#include <math.h>

// ---------- problem constants ----------
#define SEQ   2048
#define DM    1024
#define NH    16
#define DK    64
#define BATCH 2
#define ROWS  (BATCH*SEQ)          // 4096

typedef __attribute__((ext_vector_type(16))) __bf16          v16bf;
typedef __attribute__((ext_vector_type(8)))  float           v8f;
typedef __attribute__((ext_vector_type(8)))  unsigned short  v8u16;
typedef __attribute__((ext_vector_type(16))) unsigned short  v16u16;

__device__ __forceinline__ unsigned short f2bf(float f) {
    unsigned u = __float_as_uint(f);
    u = (u + 0x7FFFu + ((u >> 16) & 1u)) >> 16;
    return (unsigned short)u;
}
__device__ __forceinline__ float bf2f(unsigned short h) {
    return __uint_as_float(((unsigned)h) << 16);
}

__device__ __forceinline__ v16bf u2bf16v(v16u16 t) {
    union { v16u16 u; v16bf b; } cv; cv.u = t; return cv.b;
}

// Load one 16-bit A/B WMMA fragment: this lane's 16 bf16 values are the
// two contiguous 8-element chunks at p0 (K = 8H+0..7) and p1 (K = 16+8H+0..7).
__device__ __forceinline__ v16bf load_frag(const unsigned short* p0,
                                           const unsigned short* p1) {
    v8u16 a = *(const v8u16*)p0;
    v8u16 b = *(const v8u16*)p1;
    v16u16 t;
#pragma unroll
    for (int i = 0; i < 8; ++i) { t[i] = a[i]; t[8 + i] = b[i]; }
    return u2bf16v(t);
}

__device__ __forceinline__ v8f wmma_bf16(v16bf a, v16bf b, v8f c) {
    return __builtin_amdgcn_wmma_f32_16x16x32_bf16(
        false, a, false, b, (short)0, c, false, false);
}

// ---------- f32 -> bf16 convert ----------
__global__ void cvt_f32_bf16(const float* __restrict__ in,
                             unsigned short* __restrict__ out, int n) {
    int i = blockIdx.x * blockDim.x + threadIdx.x;
    if (i < n) out[i] = f2bf(in[i]);
}

// ---------- tiled bf16 WMMA GEMM: out[s,o] = sum_i X[s,i]*W[o,i] ----------
// mode 0: bf16 out, layout [b,h,s,dk]   (Q / K)
// mode 1: bf16 out, layout [b,h,dk,s]   (V transposed)
// mode 2: f32  out, layout [s,o] flat   (final Wo projection)
__global__ __launch_bounds__(128)
void gemm_bf16(const unsigned short* __restrict__ X,
               const unsigned short* __restrict__ W,
               unsigned short* __restrict__ outb,
               float* __restrict__ outf, int mode) {
    const int lane = threadIdx.x & 31;
    const int wid  = threadIdx.x >> 5;
    const int lm   = lane & 15;
    const int lh   = lane >> 4;
    const int m0 = blockIdx.x * 128 + (wid >> 1) * 64;   // row (s) tile
    const int n0 = blockIdx.y * 64  + (wid & 1)  * 32;   // col (o) tile

    v8f acc[4][2];
#pragma unroll
    for (int mf = 0; mf < 4; ++mf)
#pragma unroll
        for (int nf = 0; nf < 2; ++nf) acc[mf][nf] = {};

    for (int kk = 0; kk < DM; kk += 32) {
        v16bf af[4], bfr[2];
#pragma unroll
        for (int mf = 0; mf < 4; ++mf) {
            const unsigned short* p =
                X + (size_t)(m0 + mf * 16 + lm) * DM + kk + lh * 8;
            af[mf] = load_frag(p, p + 16);
        }
#pragma unroll
        for (int nf = 0; nf < 2; ++nf) {
            const unsigned short* p =
                W + (size_t)(n0 + nf * 16 + lm) * DM + kk + lh * 8;
            bfr[nf] = load_frag(p, p + 16);
        }
#pragma unroll
        for (int mf = 0; mf < 4; ++mf)
#pragma unroll
            for (int nf = 0; nf < 2; ++nf)
                acc[mf][nf] = wmma_bf16(af[mf], bfr[nf], acc[mf][nf]);
    }

#pragma unroll
    for (int mf = 0; mf < 4; ++mf)
#pragma unroll
        for (int nf = 0; nf < 2; ++nf)
#pragma unroll
            for (int r = 0; r < 8; ++r) {
                const int sg = m0 + mf * 16 + r + 8 * lh; // global row (b*SEQ+s)
                const int o  = n0 + nf * 16 + lm;         // global col
                const float v = acc[mf][nf][r];
                if (mode == 2) {
                    outf[(size_t)sg * DM + o] = v;
                } else {
                    const int b = sg >> 11, s = sg & (SEQ - 1);
                    const int h = o >> 6,  dk = o & (DK - 1);
                    size_t idx = (mode == 0)
                        ? ((size_t)(b * NH + h) * SEQ + s) * DK + dk
                        : ((size_t)(b * NH + h) * DK + dk) * SEQ + s;
                    outb[idx] = f2bf(v);
                }
            }
}

// ---------- RoPE (in place on bf16 [b,h,s,dk]) ----------
__global__ void rope_kernel(unsigned short* __restrict__ T,
                            const int* __restrict__ pos) {
    const int idx = blockIdx.x * blockDim.x + threadIdx.x;
    const int total = BATCH * NH * SEQ * (DK / 2);
    if (idx >= total) return;
    const int t  = idx & (DK / 2 - 1);
    const int s  = (idx >> 5) & (SEQ - 1);
    const int bh = idx >> 16;                 // /(32*2048)
    const size_t base = ((size_t)bh * SEQ + s) * DK + 2 * t;
    const float ang = (float)pos[s] * __powf(10000.0f, -(float)(2 * t) / (float)DK);
    const float c = __cosf(ang), sn = __sinf(ang);
    const float e = bf2f(T[base]), o = bf2f(T[base + 1]);
    T[base]     = f2bf(e * c - o * sn);
    T[base + 1] = f2bf(e * sn + o * c);
}

// ---------- causal flash attention (one 32-query tile per wave) ----------
// Q,K: bf16 [b,h,s,dk]; Vt: bf16 [b,h,dk,s]; attn out: bf16 [b,s,h*64+d]
__global__ __launch_bounds__(128)
void attn_kernel(const unsigned short* __restrict__ Q,
                 const unsigned short* __restrict__ K,
                 const unsigned short* __restrict__ Vt,
                 unsigned short* __restrict__ attn) {
    const int lane = threadIdx.x & 31;
    const int wid  = threadIdx.x >> 5;
    const int lm   = lane & 15;
    const int lh   = lane >> 4;
    const int bh   = blockIdx.y;               // b*NH + h
    const int b    = bh >> 4, h = bh & (NH - 1);
    const int q0   = blockIdx.x * 128 + wid * 32;

    const unsigned short* Qb  = Q  + (size_t)bh * SEQ * DK;
    const unsigned short* Kb  = K  + (size_t)bh * SEQ * DK;
    const unsigned short* Vtb = Vt + (size_t)bh * DK * SEQ;

    // Q^T B-fragments (N = query, K = d), loaded once
    v16bf qb[2][2];
#pragma unroll
    for (int nf = 0; nf < 2; ++nf)
#pragma unroll
        for (int kst = 0; kst < 2; ++kst) {
            const unsigned short* p =
                Qb + (size_t)(q0 + nf * 16 + lm) * DK + kst * 32 + lh * 8;
            qb[nf][kst] = load_frag(p, p + 16);
        }

    v8f ot[4][2];                 // O^T accumulators (M = d, N = q)
#pragma unroll
    for (int mf = 0; mf < 4; ++mf)
#pragma unroll
        for (int nf = 0; nf < 2; ++nf) ot[mf][nf] = {};
    float mrow[2] = { -INFINITY, -INFINITY };
    float lrow[2] = { 0.0f, 0.0f };

    const int nkb = q0 / 32 + 1;              // causal: key blocks of 32
    for (int kb = 0; kb < nkb; ++kb) {
        const int key0 = kb * 32;

        // S^T = K_blk * Q^T  (C layout: lane holds fixed q = lm)
        v8f sf[2][2];
#pragma unroll
        for (int mf = 0; mf < 2; ++mf)
#pragma unroll
            for (int nf = 0; nf < 2; ++nf) sf[mf][nf] = {};
#pragma unroll
        for (int kst = 0; kst < 2; ++kst) {
            v16bf ka[2];
#pragma unroll
            for (int mf = 0; mf < 2; ++mf) {
                const unsigned short* p =
                    Kb + (size_t)(key0 + mf * 16 + lm) * DK + kst * 32 + lh * 8;
                ka[mf] = load_frag(p, p + 16);
            }
#pragma unroll
            for (int mf = 0; mf < 2; ++mf)
#pragma unroll
                for (int nf = 0; nf < 2; ++nf)
                    sf[mf][nf] = wmma_bf16(ka[mf], qb[nf][kst], sf[mf][nf]);
        }

        // scale + causal mask (key index = key0 + mf*16 + r + 8*lh; q = q0+nf*16+lm)
#pragma unroll
        for (int mf = 0; mf < 2; ++mf)
#pragma unroll
            for (int nf = 0; nf < 2; ++nf) {
                const int q = q0 + nf * 16 + lm;
#pragma unroll
                for (int r = 0; r < 8; ++r) {
                    const int key = key0 + mf * 16 + r + 8 * lh;
                    sf[mf][nf][r] = (key <= q) ? sf[mf][nf][r] * 0.125f
                                               : -INFINITY;
                }
            }

        // online softmax (per-lane q; one shfl_xor(16) to merge key halves)
        v16bf pb[2];
        float alpha[2];
#pragma unroll
        for (int nf = 0; nf < 2; ++nf) {
            float mx = -INFINITY;
#pragma unroll
            for (int mf = 0; mf < 2; ++mf)
#pragma unroll
                for (int r = 0; r < 8; ++r) mx = fmaxf(mx, sf[mf][nf][r]);
            mx = fmaxf(mx, __shfl_xor(mx, 16, 32));
            const float mnew = fmaxf(mrow[nf], mx);
            alpha[nf] = __expf(mrow[nf] - mnew);
            float rs = 0.0f;
            v16u16 t;
#pragma unroll
            for (int mf = 0; mf < 2; ++mf)
#pragma unroll
                for (int r = 0; r < 8; ++r) {
                    const float p = __expf(sf[mf][nf][r] - mnew);
                    rs += p;
                    t[mf * 8 + r] = f2bf(p);   // exactly the P^T B-fragment
                }
            rs += __shfl_xor(rs, 16, 32);
            lrow[nf] = lrow[nf] * alpha[nf] + rs;
            mrow[nf] = mnew;
            pb[nf] = u2bf16v(t);
        }

        // rescale O^T, then O^T += V^T_blk * P^T
#pragma unroll
        for (int mf = 0; mf < 4; ++mf) {
            const unsigned short* p =
                Vtb + (size_t)(mf * 16 + lm) * SEQ + key0 + lh * 8;
            const v16bf va = load_frag(p, p + 16);
#pragma unroll
            for (int nf = 0; nf < 2; ++nf) {
                ot[mf][nf] *= alpha[nf];
                ot[mf][nf] = wmma_bf16(va, pb[nf], ot[mf][nf]);
            }
        }
    }

    // write attn[b, s, h*64 + d]; per-lane fixed s, contiguous d (16B stores)
#pragma unroll
    for (int nf = 0; nf < 2; ++nf) {
        const float inv = 1.0f / lrow[nf];
        const int s = q0 + nf * 16 + lm;
        unsigned short* dst = attn + ((size_t)(b * SEQ + s)) * DM + h * DK;
#pragma unroll
        for (int mf = 0; mf < 4; ++mf) {
            v8u16 o8;
#pragma unroll
            for (int r = 0; r < 8; ++r) o8[r] = f2bf(ot[mf][nf][r] * inv);
            *(v8u16*)(dst + mf * 16 + lh * 8) = o8;
        }
    }
}

// ---------- host-side launch ----------
extern "C" void kernel_launch(void* const* d_in, const int* in_sizes, int n_in,
                              void* d_out, int out_size, void* d_ws, size_t ws_size,
                              hipStream_t stream) {
    const float* x  = (const float*)d_in[0];
    const float* Wq = (const float*)d_in[1];
    const float* Wk = (const float*)d_in[2];
    const float* Wv = (const float*)d_in[3];
    const float* Wo = (const float*)d_in[4];
    const int* pos  = (const int*)d_in[5];

    const size_t XBN = (size_t)ROWS * DM;            // 4096*1024
    const size_t WN  = (size_t)DM * DM;              // 1024*1024
    const size_t QN  = (size_t)BATCH * NH * SEQ * DK;

    unsigned short* w   = (unsigned short*)d_ws;
    unsigned short* xb  = w;
    unsigned short* wqb = xb + XBN;
    unsigned short* wkb = wqb + WN;
    unsigned short* wvb = wkb + WN;
    unsigned short* wob = wvb + WN;
    unsigned short* Qd  = wob + WN;
    unsigned short* Kd  = Qd + QN;
    unsigned short* Vtd = Kd + QN;
    unsigned short* At  = Vtd + QN;

    // convert inputs to bf16
    cvt_f32_bf16<<<(XBN + 255) / 256, 256, 0, stream>>>(x, xb, (int)XBN);
    cvt_f32_bf16<<<(WN + 255) / 256, 256, 0, stream>>>(Wq, wqb, (int)WN);
    cvt_f32_bf16<<<(WN + 255) / 256, 256, 0, stream>>>(Wk, wkb, (int)WN);
    cvt_f32_bf16<<<(WN + 255) / 256, 256, 0, stream>>>(Wv, wvb, (int)WN);
    cvt_f32_bf16<<<(WN + 255) / 256, 256, 0, stream>>>(Wo, wob, (int)WN);

    // projections (128x64 block tile, 4 waves of 64x32 each)
    dim3 ggrid(ROWS / 128, DM / 64);
    gemm_bf16<<<ggrid, 128, 0, stream>>>(xb, wqb, Qd, nullptr, 0);
    gemm_bf16<<<ggrid, 128, 0, stream>>>(xb, wkb, Kd, nullptr, 0);
    gemm_bf16<<<ggrid, 128, 0, stream>>>(xb, wvb, Vtd, nullptr, 1);

    // RoPE on Q and K
    const int rn = BATCH * NH * SEQ * (DK / 2);
    rope_kernel<<<(rn + 255) / 256, 256, 0, stream>>>(Qd, pos);
    rope_kernel<<<(rn + 255) / 256, 256, 0, stream>>>(Kd, pos);

    // causal flash attention
    attn_kernel<<<dim3(SEQ / 128, BATCH * NH), 128, 0, stream>>>(Qd, Kd, Vtd, At);

    // output projection -> f32
    gemm_bf16<<<ggrid, 128, 0, stream>>>(At, wob, nullptr, (float*)d_out, 2);
}
